// GCN2_63780264346290
// MI455X (gfx1250) — matrix operand order repaired and checked by reading, not compiled, not run on hardware
//
#include <hip/hip_runtime.h>
#include <hip/hip_bf16.h>
#include <math.h>

typedef __attribute__((ext_vector_type(16))) _Float16 v16h;
typedef __attribute__((ext_vector_type(8)))  float    v8f;

#define F_IN 128
#define HID  30
#define MID  4
#define NCLS 10

// ---------------------------------------------------------------------------
// fill: grid-stride constant fill (init deg=1.0 for self-loop, aggs=0)
// ---------------------------------------------------------------------------
__global__ void fill_kernel(float* __restrict__ p, float v, long long n) {
    long long i = (long long)blockIdx.x * blockDim.x + threadIdx.x;
    long long stride = (long long)gridDim.x * blockDim.x;
    for (; i < n; i += stride) p[i] = v;
}

// ---------------------------------------------------------------------------
// in-degree (incl. self-loop via deg init = 1.0)
// ---------------------------------------------------------------------------
__global__ void deg_count_kernel(const int* __restrict__ col,
                                 float* __restrict__ deg, int E) {
    int e = blockIdx.x * blockDim.x + threadIdx.x;
    __builtin_prefetch(col + e + 4096, 0, 1);   // global_prefetch_b8
    if (e < E) atomicAdd(&deg[col[e]], 1.0f);
}

__global__ void dis_kernel(const float* __restrict__ deg,
                           float* __restrict__ dis, int n) {
    int i = blockIdx.x * blockDim.x + threadIdx.x;
    if (i < n) dis[i] = rsqrtf(deg[i]);        // deg >= 1 always
}

// ---------------------------------------------------------------------------
// h1 = x @ W1   [N,128] x [128,30] -> [N,30]   via v_wmma_f32_16x16x32_f16
// W1 staged in LDS transposed + f16-converted + zero-padded to 32 cols, so
// each B fragment is one contiguous 32-byte ds read (no divergence in loop).
// One wave per 16-row tile; 8 waves (256 threads) per block.
// ---------------------------------------------------------------------------
__global__ void gemm1_wmma_kernel(const float* __restrict__ x,
                                  const float* __restrict__ W1,
                                  float* __restrict__ h1, int n_nodes) {
    __shared__ __align__(32) _Float16 sWt[32 * F_IN];   // [col][k], 8 KB

    const int t = threadIdx.x;
    for (int idx = t; idx < 32 * F_IN; idx += blockDim.x) {
        int c = idx >> 7;       // col 0..31
        int k = idx & 127;      // k   0..127
        sWt[idx] = (c < HID) ? (_Float16)W1[(size_t)k * HID + c] : (_Float16)0.0f;
    }
    __syncthreads();

    const int wave  = t >> 5;
    const int lane  = t & 31;
    const int mtile = blockIdx.x * (blockDim.x >> 5) + wave;
    const int m0    = mtile * 16;
    if (m0 >= n_nodes) return;                 // wave-uniform exit (EXEC stays all-1)

    const int m  = lane & 15;
    const int hi = lane >> 4;
    const int n  = lane & 15;

    int rowidx = m0 + m;
    if (rowidx >= n_nodes) rowidx = n_nodes - 1;   // branchless clamp (no tail for N=100000)
    const float* xrow = x + (size_t)rowidx * F_IN;

    v8f acc0 = {};   // output cols 0..15
    v8f acc1 = {};   // output cols 16..29 (+2 pad)

    #pragma unroll
    for (int k0 = 0; k0 < F_IN; k0 += 32) {
        // A fragment: 16x32 f16, contiguous K order k = 16*hi + s (matches B);
        // one contiguous 16-float run of this lane's x row -> b128 loads + cvt_pk
        v16h a;
        #pragma unroll
        for (int j = 0; j < 8; ++j) {
            int kb = k0 + 16 * hi + 2 * j;
            a[2 * j]     = (_Float16)xrow[kb];
            a[2 * j + 1] = (_Float16)xrow[kb + 1];
        }
        // B fragments: lane needs W[k0+16*hi .. +15][col] == contiguous run
        // in the transposed LDS tile
        v16h b0 = *(const v16h*)&sWt[(size_t)n * F_IN + k0 + 16 * hi];
        v16h b1 = *(const v16h*)&sWt[(size_t)(n + 16) * F_IN + k0 + 16 * hi];

        acc0 = __builtin_amdgcn_wmma_f32_16x16x32_f16(false, a, false, b0,
                                                      (short)0, acc0, false, false);
        acc1 = __builtin_amdgcn_wmma_f32_16x16x32_f16(false, a, false, b1,
                                                      (short)0, acc1, false, false);
    }

    // D layout: VGPR v -> row m0 + v + 8*hi, col = lane&15
    #pragma unroll
    for (int v = 0; v < 8; ++v) {
        int rm = m0 + v + 8 * hi;
        if (rm < n_nodes) {
            h1[(size_t)rm * HID + n] = acc0[v];
            int c1 = 16 + n;
            if (c1 < HID) h1[(size_t)rm * HID + c1] = acc1[v];
        }
    }
}

// ---------------------------------------------------------------------------
// scatter-add, 30 features: one wave per edge, lane = feature
// ---------------------------------------------------------------------------
__global__ void scatter30_kernel(const int* __restrict__ row,
                                 const int* __restrict__ col,
                                 const float* __restrict__ dis,
                                 const float* __restrict__ h,
                                 float* __restrict__ agg, int E) {
    int e = blockIdx.x * (blockDim.x >> 5) + (threadIdx.x >> 5);
    if (e >= E) return;
    int lane = threadIdx.x & 31;
    if (lane == 0) {
        __builtin_prefetch(row + e + 1024, 0, 1);
        __builtin_prefetch(col + e + 1024, 0, 1);
    }
    int r = row[e], c = col[e];
    float w = dis[r] * dis[c];
    if (lane < HID)
        atomicAdd(&agg[(size_t)c * HID + lane], w * h[(size_t)r * HID + lane]);
}

// ---------------------------------------------------------------------------
// scatter-add, 4 features: one thread per edge (float4 gather)
// ---------------------------------------------------------------------------
__global__ void scatter4_kernel(const int* __restrict__ row,
                                const int* __restrict__ col,
                                const float* __restrict__ dis,
                                const float* __restrict__ h,
                                float* __restrict__ agg, int E) {
    int e = blockIdx.x * blockDim.x + threadIdx.x;
    __builtin_prefetch(row + e + 4096, 0, 1);
    if (e >= E) return;
    int r = row[e], c = col[e];
    float w = dis[r] * dis[c];
    const float4 hv = *(const float4*)(h + (size_t)r * MID);
    float* a = agg + (size_t)c * MID;
    atomicAdd(a + 0, w * hv.x);
    atomicAdd(a + 1, w * hv.y);
    atomicAdd(a + 2, w * hv.z);
    atomicAdd(a + 3, w * hv.w);
}

// ---------------------------------------------------------------------------
// finalize: add self-loop term h/deg, optional mean, bias, relu (in place into agg)
// ---------------------------------------------------------------------------
__global__ void finalize_kernel(const float* __restrict__ h,
                                float* __restrict__ agg,
                                const float* __restrict__ deg,
                                const float* __restrict__ b,
                                int n, int F, int mean_aggr) {
    long long i = (long long)blockIdx.x * blockDim.x + threadIdx.x;
    long long total = (long long)n * F;
    if (i >= total) return;
    int node = (int)(i / F);
    int f    = (int)(i % F);
    float dg = deg[node];
    float v  = agg[i] + h[i] / dg;     // self-loop: norm = dis^2 = 1/deg
    if (mean_aggr) v /= dg;            // deg >= 1 so max(deg,1)==deg
    agg[i] = fmaxf(v + b[f], 0.0f);
}

// ---------------------------------------------------------------------------
// small dense GEMM per node (Fin,Fout tiny); W staged in LDS
// ---------------------------------------------------------------------------
__global__ void dense_small_kernel(const float* __restrict__ in,
                                   const float* __restrict__ W,
                                   float* __restrict__ out,
                                   int n, int Fin, int Fout) {
    __shared__ float sW[512];
    int t = threadIdx.x;
    for (int k = t; k < Fin * Fout; k += blockDim.x) sW[k] = W[k];
    __syncthreads();
    int i = blockIdx.x * blockDim.x + t;
    if (i >= n) return;
    const float* xi = in + (size_t)i * Fin;
    for (int c = 0; c < Fout; ++c) {
        float acc = 0.f;
        for (int k = 0; k < Fin; ++k) acc = fmaf(xi[k], sW[k * Fout + c], acc);
        out[(size_t)i * Fout + c] = acc;
    }
}

// ---------------------------------------------------------------------------
// classifier: logits = h @ Wc + bc (4->10), then log_softmax
// ---------------------------------------------------------------------------
__global__ void classify_kernel(const float* __restrict__ h,
                                const float* __restrict__ Wc,
                                const float* __restrict__ bc,
                                float* __restrict__ out, int n) {
    __shared__ float sW[MID * NCLS];
    __shared__ float sb[NCLS];
    int t = threadIdx.x;
    if (t < MID * NCLS) sW[t] = Wc[t];
    if (t < NCLS) sb[t] = bc[t];
    __syncthreads();
    int i = blockIdx.x * blockDim.x + t;
    if (i >= n) return;
    float h0 = h[(size_t)i * MID + 0], h1 = h[(size_t)i * MID + 1];
    float h2 = h[(size_t)i * MID + 2], h3 = h[(size_t)i * MID + 3];
    float l[NCLS];
    float m = -1e30f;
    #pragma unroll
    for (int c = 0; c < NCLS; ++c) {
        float v = h0 * sW[0 * NCLS + c] + h1 * sW[1 * NCLS + c]
                + h2 * sW[2 * NCLS + c] + h3 * sW[3 * NCLS + c] + sb[c];
        l[c] = v;
        m = fmaxf(m, v);
    }
    float s = 0.f;
    #pragma unroll
    for (int c = 0; c < NCLS; ++c) s += expf(l[c] - m);
    float lse = m + logf(s);
    #pragma unroll
    for (int c = 0; c < NCLS; ++c) out[(size_t)i * NCLS + c] = l[c] - lse;
}

// ---------------------------------------------------------------------------
extern "C" void kernel_launch(void* const* d_in, const int* in_sizes, int n_in,
                              void* d_out, int out_size, void* d_ws, size_t ws_size,
                              hipStream_t stream) {
    const float* x  = (const float*)d_in[0];
    const int*   ei = (const int*)d_in[1];    // [2,E] (row; col)
    const float* W1 = (const float*)d_in[2];
    const float* b1 = (const float*)d_in[3];
    const float* W2 = (const float*)d_in[4];
    const float* b2 = (const float*)d_in[5];
    const float* W3 = (const float*)d_in[6];
    const float* b3 = (const float*)d_in[7];
    const float* Wc = (const float*)d_in[8];
    const float* bc = (const float*)d_in[9];

    const int N = in_sizes[0] / F_IN;
    const int E = in_sizes[1] / 2;
    const int* row = ei;
    const int* col = ei + E;

    // workspace layout (floats): deg | dis | h1(30N) | h2(4N) | h3(4N) | agg1(30N) | agg2(4N) | agg3(4N)
    float* ws   = (float*)d_ws;
    float* deg  = ws;
    float* dis  = deg + N;
    float* h1   = dis + N;
    float* h2   = h1 + (size_t)30 * N;
    float* h3   = h2 + (size_t)4 * N;
    float* agg1 = h3 + (size_t)4 * N;
    float* agg2 = agg1 + (size_t)30 * N;
    float* agg3 = agg2 + (size_t)4 * N;

    const int TB = 256;

    // init: deg = 1 (self-loop), all agg buffers = 0 (contiguous 38N floats)
    fill_kernel<<<1024, TB, 0, stream>>>(deg, 1.0f, (long long)N);
    fill_kernel<<<2048, TB, 0, stream>>>(agg1, 0.0f, (long long)38 * N);

    // degree + symmetric norm
    deg_count_kernel<<<(E + TB - 1) / TB, TB, 0, stream>>>(col, deg, E);
    dis_kernel<<<(N + TB - 1) / TB, TB, 0, stream>>>(deg, dis, N);

    // ---- layer 1: 128 -> 30, mean aggr ----
    {
        int mtiles = (N + 15) / 16;
        int wavesPerBlock = 8;                 // blockDim 256 = 8 waves
        int blocks = (mtiles + wavesPerBlock - 1) / wavesPerBlock;
        gemm1_wmma_kernel<<<blocks, 32 * wavesPerBlock, 0, stream>>>(x, W1, h1, N);
    }
    {
        int wavesPerBlock = TB / 32;
        int blocks = (E + wavesPerBlock - 1) / wavesPerBlock;
        scatter30_kernel<<<blocks, TB, 0, stream>>>(row, col, dis, h1, agg1, E);
    }
    {
        long long total = (long long)N * HID;
        finalize_kernel<<<(int)((total + TB - 1) / TB), TB, 0, stream>>>(h1, agg1, deg, b1, N, HID, 1);
    }

    // ---- layer 2: 30 -> 4, mean aggr ----
    dense_small_kernel<<<(N + TB - 1) / TB, TB, 0, stream>>>(agg1, W2, h2, N, HID, MID);
    scatter4_kernel<<<(E + TB - 1) / TB, TB, 0, stream>>>(row, col, dis, h2, agg2, E);
    {
        long long total = (long long)N * MID;
        finalize_kernel<<<(int)((total + TB - 1) / TB), TB, 0, stream>>>(h2, agg2, deg, b2, N, MID, 1);
    }

    // ---- layer 3: 4 -> 4, sum aggr ----
    dense_small_kernel<<<(N + TB - 1) / TB, TB, 0, stream>>>(agg2, W3, h3, N, MID, MID);
    scatter4_kernel<<<(E + TB - 1) / TB, TB, 0, stream>>>(row, col, dis, h3, agg3, E);
    {
        long long total = (long long)N * MID;
        finalize_kernel<<<(int)((total + TB - 1) / TB), TB, 0, stream>>>(h3, agg3, deg, b3, N, MID, 0);
    }

    // ---- classifier + log_softmax ----
    classify_kernel<<<(N + TB - 1) / TB, TB, 0, stream>>>(agg3, Wc, bc, (float*)d_out, N);
}
